// SparseMoERouter_87875030876714
// MI455X (gfx1250) — compile-verified, matrix operand order checked by reading, not run on hardware
//
#include <hip/hip_runtime.h>
#include <hip/hip_bf16.h>

#define T_TOK 8192
#define DIM   1024
#define NEXP  8
#define HID   4096

#define BM 128
#define BN 128
#define BK 32

#if defined(__has_builtin)
# if __has_builtin(__builtin_amdgcn_global_load_async_to_lds_b128) && \
     __has_builtin(__builtin_amdgcn_s_wait_asynccnt)
#  define HAVE_ASYNC 1
# endif
#endif
#ifndef HAVE_ASYNC
# define HAVE_ASYNC 0
#endif

typedef __attribute__((ext_vector_type(16))) __bf16 v16bf;
typedef __attribute__((ext_vector_type(8)))  float  v8f;
typedef __attribute__((ext_vector_type(8)))  unsigned short ushort8;
typedef __attribute__((ext_vector_type(4)))  int v4i;

#if HAVE_ASYNC
typedef __attribute__((address_space(1))) v4i* gptr_v4i;
typedef __attribute__((address_space(3))) v4i* lptr_v4i;
#endif

__device__ __forceinline__ unsigned short f2bf(float f) {
    unsigned u = __float_as_uint(f);
    unsigned r = u + 0x7fffu + ((u >> 16) & 1u);
    return (unsigned short)(r >> 16);
}

__device__ __forceinline__ v16bf ldfrag(const unsigned short* p0, const unsigned short* p1) {
    union { v16bf v; ushort8 h[2]; } u;
    u.h[0] = *(const ushort8*)p0;
    u.h[1] = *(const ushort8*)p1;
    return u.v;
}

// Copy 16B global -> LDS, async (ASYNCcnt-tracked) when available.
__device__ __forceinline__ void cp16(unsigned short* lds_dst, const unsigned short* gsrc) {
#if HAVE_ASYNC
    __builtin_amdgcn_global_load_async_to_lds_b128(
        (gptr_v4i)gsrc, (lptr_v4i)lds_dst, 0, 0);
#else
    *(ushort8*)lds_dst = *(const ushort8*)gsrc;
#endif
}

__device__ __forceinline__ void cp_fence() {
#if HAVE_ASYNC
    __builtin_amdgcn_s_wait_asynccnt(0);
#endif
}

// ---------------------------------------------------------------- zero init
__global__ void zero_init_kernel(float* __restrict__ out, unsigned* __restrict__ stats) {
    size_t idx = (size_t)blockIdx.x * blockDim.x + threadIdx.x;
    if (idx < (size_t)T_TOK * DIM) out[idx] = 0.0f;
    if (blockIdx.x == 0 && threadIdx.x < 128) stats[threadIdx.x] = 0u;  // 512B stats region
}

// ---------------------------------------------------------------- x -> bf16
__global__ void convert_x_kernel(const float* __restrict__ x, unsigned short* __restrict__ xbf) {
    size_t idx = (size_t)blockIdx.x * blockDim.x + threadIdx.x;
    if (idx < (size_t)T_TOK * DIM) xbf[idx] = f2bf(x[idx]);
}

// ------------------------------------------- W [e][R][C] f32 -> [e][C][R] bf16
__global__ __launch_bounds__(256)
void transpose_convert_kernel(const float* __restrict__ in, unsigned short* __restrict__ out,
                              int R, int C) {
    __shared__ float tile[64][65];
    const float* inp = in + (size_t)blockIdx.z * R * C;
    unsigned short* outp = out + (size_t)blockIdx.z * R * C;
    int c0 = blockIdx.x * 64, r0 = blockIdx.y * 64;
    int tx = threadIdx.x & 63, ty = threadIdx.x >> 6;   // 64 x 4 threads
    #pragma unroll
    for (int r = ty; r < 64; r += 4)
        tile[r][tx] = inp[(size_t)(r0 + r) * C + c0 + tx];
    __syncthreads();
    #pragma unroll
    for (int r = ty; r < 64; r += 4)
        outp[(size_t)(c0 + r) * R + r0 + tx] = f2bf(tile[tx][r]);
}

// ---------------------------------------------------------------- router
__global__ __launch_bounds__(256)
void router_kernel(const float* __restrict__ x, const float* __restrict__ Wr,
                   unsigned* __restrict__ cnt, float* __restrict__ usage,
                   float* __restrict__ sel, float* __restrict__ zsum,
                   int* __restrict__ toklist, float* __restrict__ wlist) {
    __shared__ float sWr[DIM * NEXP];   // 32 KB
    __shared__ float sUsage[NEXP];
    __shared__ float sSel[NEXP];
    __shared__ float sZ;
    int tid = threadIdx.x;
    for (int i = tid; i < DIM * NEXP; i += 256) sWr[i] = Wr[i];
    if (tid < NEXP) { sUsage[tid] = 0.0f; sSel[tid] = 0.0f; }
    if (tid == NEXP) sZ = 0.0f;
    __syncthreads();

    int wave = tid >> 5, lane = tid & 31;
    int t = blockIdx.x * 8 + wave;
    const float* xr = x + (size_t)t * DIM;

    float acc[NEXP];
    #pragma unroll
    for (int e = 0; e < NEXP; e++) acc[e] = 0.0f;
    for (int d = lane; d < DIM; d += 32) {
        float xv = xr[d];
        #pragma unroll
        for (int e = 0; e < NEXP; e++) acc[e] += xv * sWr[d * NEXP + e];
    }
    #pragma unroll
    for (int off = 16; off > 0; off >>= 1) {
        #pragma unroll
        for (int e = 0; e < NEXP; e++) acc[e] += __shfl_down(acc[e], off);
    }

    if (lane == 0) {
        float mx = acc[0];
        #pragma unroll
        for (int e = 1; e < NEXP; e++) mx = fmaxf(mx, acc[e]);
        float p[NEXP], s = 0.0f;
        #pragma unroll
        for (int e = 0; e < NEXP; e++) { p[e] = expf(acc[e] - mx); s += p[e]; }
        float inv = 1.0f / s;
        float z = mx + logf(s);

        int e0 = 0;
        #pragma unroll
        for (int e = 1; e < NEXP; e++) if (p[e] > p[e0]) e0 = e;
        int e1 = (e0 == 0) ? 1 : 0;
        #pragma unroll
        for (int e = 0; e < NEXP; e++) if (e != e0 && p[e] > p[e1]) e1 = e;

        float p0 = p[e0] * inv, p1 = p[e1] * inv;
        float wn = 1.0f / (p0 + p1);
        float w0 = p0 * wn, w1 = p1 * wn;

        // LDS (ds_add_f32) accumulation, flushed once per block
        #pragma unroll
        for (int e = 0; e < NEXP; e++) atomicAdd(&sUsage[e], p[e] * inv);
        atomicAdd(&sSel[e0], 1.0f);
        atomicAdd(&sSel[e1], 1.0f);
        atomicAdd(&sZ, z * z);

        unsigned pos0 = atomicAdd(&cnt[e0], 1u);
        toklist[e0 * T_TOK + pos0] = t;
        wlist[e0 * T_TOK + pos0] = w0;
        unsigned pos1 = atomicAdd(&cnt[e1], 1u);
        toklist[e1 * T_TOK + pos1] = t;
        wlist[e1 * T_TOK + pos1] = w1;
    }
    __syncthreads();
    if (tid < NEXP) {
        atomicAdd(&usage[tid], sUsage[tid]);
        atomicAdd(&sel[tid], sSel[tid]);
    }
    if (tid == NEXP) atomicAdd(zsum, sZ);
}

// ---------------------------------------------------------------- prefix
__global__ void prefix_kernel(const unsigned* __restrict__ cnt, unsigned* __restrict__ offs) {
    if (threadIdx.x == 0) {
        unsigned a = 0;
        for (int e = 0; e < NEXP; e++) { offs[e] = a; a += cnt[e]; }
    }
}

// --------------------------------- GEMM1: h = gelu(xg @ W1[e] + b1[e]), bf16
// Block tile 128x128x32, 8 waves in 4(M)x2(N); each wave: 32x64 = 8 WMMA/k-step.
__global__ __launch_bounds__(256)
void gemm1_kernel(const unsigned short* __restrict__ xbf,
                  const unsigned short* __restrict__ w1t,   // [E][H][D]
                  const float* __restrict__ b1,
                  unsigned short* __restrict__ hbuf,        // [2T][H]
                  const unsigned* __restrict__ cnt, const unsigned* __restrict__ offs,
                  const int* __restrict__ toklist) {
    int e = blockIdx.z;
    unsigned n_e = cnt[e];
    unsigned m0 = blockIdx.y * BM;
    if (m0 >= n_e) return;
    int n0 = blockIdx.x * BN;

    __shared__ __align__(16) unsigned short As[BM][BK];
    __shared__ __align__(16) unsigned short Bs[BN][BK];
    __shared__ int sTok[BM];

    int tid = threadIdx.x;
    if (tid < BM) {
        unsigned r = m0 + (unsigned)tid;
        sTok[tid] = toklist[e * T_TOK + (r < n_e ? r : 0u)];
    }
    __syncthreads();

    int lane = tid & 31, wave = tid >> 5;
    int wm = wave & 3, wn = wave >> 2;
    int mrow = lane & 15;
    int koff = (lane < 16) ? 0 : 8;

    v8f acc[2][4] = {};

    for (int k0 = 0; k0 < DIM; k0 += BK) {
        __syncthreads();
        #pragma unroll
        for (int it = 0; it < 2; it++) {          // A: 128x32 bf16 = 512 x 16B
            int j = tid + 256 * it;
            int r = j >> 2, seg = (j & 3) * 8;
            cp16(&As[r][seg], xbf + (size_t)sTok[r] * DIM + k0 + seg);
        }
        #pragma unroll
        for (int it = 0; it < 2; it++) {          // B: 128x32 bf16 = 512 x 16B
            int j = tid + 256 * it;
            int r = j >> 2, seg = (j & 3) * 8;
            cp16(&Bs[r][seg], w1t + ((size_t)e * HID + n0 + r) * DIM + k0 + seg);
        }
        cp_fence();
        __syncthreads();

        v16bf a0 = ldfrag(&As[wm * 32 + mrow][koff],      &As[wm * 32 + mrow][koff + 16]);
        v16bf a1 = ldfrag(&As[wm * 32 + 16 + mrow][koff], &As[wm * 32 + 16 + mrow][koff + 16]);
        #pragma unroll
        for (int ns = 0; ns < 4; ns++) {
            v16bf bf = ldfrag(&Bs[wn * 64 + ns * 16 + mrow][koff],
                              &Bs[wn * 64 + ns * 16 + mrow][koff + 16]);
            acc[0][ns] = __builtin_amdgcn_wmma_f32_16x16x32_bf16(
                false, a0, false, bf, (short)0, acc[0][ns], false, false);
            acc[1][ns] = __builtin_amdgcn_wmma_f32_16x16x32_bf16(
                false, a1, false, bf, (short)0, acc[1][ns], false, false);
        }
    }

    unsigned hbase = offs[e];
    #pragma unroll
    for (int sub = 0; sub < 2; sub++) {
        #pragma unroll
        for (int ns = 0; ns < 4; ns++) {
            int n = n0 + wn * 64 + ns * 16 + (lane & 15);
            float bias = b1[e * HID + n];
            #pragma unroll
            for (int v = 0; v < 8; v++) {
                int m = v + ((lane >> 4) << 3);         // 0..15
                unsigned r = m0 + (unsigned)(wm * 32 + sub * 16 + m);
                if (r < n_e) {
                    float val = acc[sub][ns][v] + bias;
                    float g = 0.5f * val * (1.0f + erff(val * 0.70710678118f));
                    hbuf[(size_t)(hbase + r) * HID + n] = f2bf(g);
                }
            }
        }
    }
}

// --------------------------------- GEMM2: out[tok] += w * (h @ W2[e] + b2[e])
__global__ __launch_bounds__(256)
void gemm2_kernel(const unsigned short* __restrict__ hbuf,  // [2T][H]
                  const unsigned short* __restrict__ w2t,   // [E][D][H]
                  const float* __restrict__ b2,
                  float* __restrict__ out,
                  const unsigned* __restrict__ cnt, const unsigned* __restrict__ offs,
                  const int* __restrict__ toklist, const float* __restrict__ wlist) {
    int e = blockIdx.z;
    unsigned n_e = cnt[e];
    unsigned m0 = blockIdx.y * BM;
    if (m0 >= n_e) return;
    int n0 = blockIdx.x * BN;

    __shared__ __align__(16) unsigned short As[BM][BK];
    __shared__ __align__(16) unsigned short Bs[BN][BK];
    __shared__ int sTok[BM];
    __shared__ float sW[BM];

    int tid = threadIdx.x;
    if (tid < BM) {
        unsigned r = m0 + (unsigned)tid;
        unsigned rc = (r < n_e) ? r : 0u;
        sTok[tid] = toklist[e * T_TOK + rc];
        sW[tid] = (r < n_e) ? wlist[e * T_TOK + rc] : 0.0f;
    }
    __syncthreads();

    int lane = tid & 31, wave = tid >> 5;
    int wm = wave & 3, wn = wave >> 2;
    int mrow = lane & 15;
    int koff = (lane < 16) ? 0 : 8;
    unsigned hbase = offs[e];

    v8f acc[2][4] = {};

    for (int k0 = 0; k0 < HID; k0 += BK) {
        __syncthreads();
        #pragma unroll
        for (int it = 0; it < 2; it++) {
            int j = tid + 256 * it;
            int r = j >> 2, seg = (j & 3) * 8;
            unsigned rr = m0 + (unsigned)r;
            if (rr >= n_e) rr = n_e - 1;
            cp16(&As[r][seg], hbuf + (size_t)(hbase + rr) * HID + k0 + seg);
        }
        #pragma unroll
        for (int it = 0; it < 2; it++) {
            int j = tid + 256 * it;
            int r = j >> 2, seg = (j & 3) * 8;
            cp16(&Bs[r][seg], w2t + ((size_t)e * DIM + n0 + r) * HID + k0 + seg);
        }
        cp_fence();
        __syncthreads();

        v16bf a0 = ldfrag(&As[wm * 32 + mrow][koff],      &As[wm * 32 + mrow][koff + 16]);
        v16bf a1 = ldfrag(&As[wm * 32 + 16 + mrow][koff], &As[wm * 32 + 16 + mrow][koff + 16]);
        #pragma unroll
        for (int ns = 0; ns < 4; ns++) {
            v16bf bf = ldfrag(&Bs[wn * 64 + ns * 16 + mrow][koff],
                              &Bs[wn * 64 + ns * 16 + mrow][koff + 16]);
            acc[0][ns] = __builtin_amdgcn_wmma_f32_16x16x32_bf16(
                false, a0, false, bf, (short)0, acc[0][ns], false, false);
            acc[1][ns] = __builtin_amdgcn_wmma_f32_16x16x32_bf16(
                false, a1, false, bf, (short)0, acc[1][ns], false, false);
        }
    }

    #pragma unroll
    for (int sub = 0; sub < 2; sub++) {
        #pragma unroll
        for (int ns = 0; ns < 4; ns++) {
            int n = n0 + wn * 64 + ns * 16 + (lane & 15);
            float bias = b2[e * DIM + n];
            #pragma unroll
            for (int v = 0; v < 8; v++) {
                int m = v + ((lane >> 4) << 3);
                int rloc = wm * 32 + sub * 16 + m;
                unsigned r = m0 + (unsigned)rloc;
                if (r < n_e) {
                    float val = (acc[sub][ns][v] + bias) * sW[rloc];
                    atomicAdd(&out[(size_t)sTok[rloc] * DIM + n], val);
                }
            }
        }
    }
}

// ---------------------------------------------------------------- aux finalize
__global__ void finalize_kernel(const float* __restrict__ usage, const float* __restrict__ sel,
                                const float* __restrict__ zsum, float* __restrict__ out_aux) {
    if (threadIdx.x == 0) {
        float lb = 0.0f;
        for (int e = 0; e < NEXP; e++)
            lb += (usage[e] / (float)T_TOK) * (sel[e] / (float)(T_TOK * 2));
        lb *= (float)NEXP;
        float zl = zsum[0] / (float)T_TOK;
        out_aux[0] = 0.01f * lb + 0.01f * zl;
    }
}

// ---------------------------------------------------------------- launch
extern "C" void kernel_launch(void* const* d_in, const int* in_sizes, int n_in,
                              void* d_out, int out_size, void* d_ws, size_t ws_size,
                              hipStream_t stream) {
    const float* x  = (const float*)d_in[0];
    const float* Wr = (const float*)d_in[1];
    const float* W1 = (const float*)d_in[2];
    const float* b1 = (const float*)d_in[3];
    const float* W2 = (const float*)d_in[4];
    const float* b2 = (const float*)d_in[5];
    float* out = (float*)d_out;

    unsigned char* ws = (unsigned char*)d_ws;
    unsigned* cnt   = (unsigned*)(ws + 0);
    unsigned* offs  = (unsigned*)(ws + 32);
    float*    usage = (float*)(ws + 64);
    float*    sel   = (float*)(ws + 96);
    float*    zsum  = (float*)(ws + 128);
    int*      toklist = (int*)(ws + 512);
    float*    wlist   = (float*)(ws + 512 + (size_t)NEXP * T_TOK * 4);
    size_t big = (size_t)1 << 20;
    unsigned short* xbf  = (unsigned short*)(ws + big);                         // 16 MB
    unsigned short* w1t  = (unsigned short*)(ws + big + 16 * big);              // 64 MB
    unsigned short* w2t  = (unsigned short*)(ws + big + 80 * big);              // 64 MB
    unsigned short* hbuf = (unsigned short*)(ws + big + 144 * big);             // 128 MB

    size_t nout = (size_t)T_TOK * DIM;
    zero_init_kernel<<<(unsigned)((nout + 255) / 256), 256, 0, stream>>>(out, cnt);
    convert_x_kernel<<<(unsigned)((nout + 255) / 256), 256, 0, stream>>>(x, xbf);

    dim3 gT1(HID / 64, DIM / 64, NEXP);   // W1: [E][D][H] -> [E][H][D]
    transpose_convert_kernel<<<gT1, 256, 0, stream>>>(W1, w1t, DIM, HID);
    dim3 gT2(DIM / 64, HID / 64, NEXP);   // W2: [E][H][D] -> [E][D][H]
    transpose_convert_kernel<<<gT2, 256, 0, stream>>>(W2, w2t, HID, DIM);

    router_kernel<<<T_TOK / 8, 256, 0, stream>>>(x, Wr, cnt, usage, sel, zsum, toklist, wlist);
    prefix_kernel<<<1, 32, 0, stream>>>(cnt, offs);

    dim3 g1(HID / BN, T_TOK / BM, NEXP);
    gemm1_kernel<<<g1, 256, 0, stream>>>(xbf, w1t, b1, hbuf, cnt, offs, toklist);

    dim3 g2(DIM / BN, T_TOK / BM, NEXP);
    gemm2_kernel<<<g2, 256, 0, stream>>>(hbuf, w2t, b2, out, cnt, offs, toklist, wlist);

    finalize_kernel<<<1, 32, 0, stream>>>(usage, sel, zsum, out + nout);
}
